// ConformerEncoder_4483945857026
// MI455X (gfx1250) — matrix operand order repaired and checked by reading, not compiled, not run on hardware
//
#include <hip/hip_runtime.h>

constexpr int kC = 256, kH = 2, kHS = 128, kMID = 1024, kL = 6, kB = 8, kT = 768;
constexpr int kN  = kB * kT;        // 6144 columns in the (b,t) flattening
constexpr int kS2 = 2 * kT - 1;     // 1535 relative positions
constexpr float kEPS = 1e-5f;

typedef __attribute__((ext_vector_type(16))) _Float16 v16h;
typedef __attribute__((ext_vector_type(8)))  float    v8f;

// ---------------------------------------------------------------------------
// Batched WMMA GEMM:  C = alpha*(A@B) + bias_m + beta*C + add_scale*S
// Template-specialized data paths (no per-element runtime branching):
//   A16K   : A is f16 with lda_k==1 (weights / probs). Else A is f32 with
//            lda_m==1 (q-path) and optional per-K bias (bias_u / bias_v).
//   BNCONT : B is f32 with ldb_n==1 (activations). Else f32 with ldb_k==1 (v).
//   COMP   : N axis is flattened (batch,time); tile width 256 divides T, so
//            batch index and time base are block-uniform. Supports temporal
//            shift with zero boundary (k=3 / k=7 convs as shifted GEMMs).
//   BRELU  : ReLU applied to B on load (fuses FFN activation).
// Block: 256 threads = 8 waves; block tile 64(M) x 256(N);
// wave tile 32x64 = 2x4 grid of v_wmma_f32_16x16x32_f16.
// ---------------------------------------------------------------------------
struct GP {
  const void* Aptr; const void* Bptr; float* Cptr;
  const float* abias; const float* bias_m; const float* add_src;
  long lda_m, lda_k;
  long ldb_k, ldb_n, ldb_bat;
  long ldc_m, ldc_n, ldc_bat;
  long add_ldm, add_ldn, add_bat;
  long batA_b, batA_h, batB_b, batB_h, batC_b, batC_h;
  long abias_b, abias_h;
  int M, N, K, nbh;
  int shift, Tperiod;
  float alpha, beta, add_scale;
};

template <bool A16K, bool BNCONT, bool COMP, bool BRELU>
__global__ __launch_bounds__(256) void gemm_wmma_kernel(GP p) {
  __shared__ _Float16 As[64 * 32];     // [m][k]
  __shared__ _Float16 Bs[256 * 32];    // [n][k]  (transposed for dword frags)

  const int tid  = threadIdx.x;
  const int lane = tid & 31;
  const int wave = tid >> 5;
  const int half = lane >> 4;
  const int lr   = lane & 15;
  const int wm   = wave & 1;           // 2 waves along M
  const int wn   = wave >> 1;          // 4 waves along N

  const long bm = (long)blockIdx.y * 64;
  const long bn = (long)blockIdx.x * 256;
  const int  z  = blockIdx.z;
  const int  zb = z / p.nbh;
  const int  zh = z % p.nbh;

  const long aoff0 = (long)zb * p.batA_b + (long)zh * p.batA_h;
  const long boff0 = (long)zb * p.batB_b + (long)zh * p.batB_h;
  const long coff0 = (long)zb * p.batC_b + (long)zh * p.batC_h;
  const float* abp = nullptr;
  if (!A16K) abp = p.abias ? p.abias + (long)zb * p.abias_b + (long)zh * p.abias_h
                           : nullptr;

  // block-uniform composite-N decomposition (256 | Tperiod by construction)
  long bi = 0, nt0 = bn;
  if (COMP) { bi = bn / p.Tperiod; nt0 = bn - bi * p.Tperiod; }

  v8f zerov = {0.f, 0.f, 0.f, 0.f, 0.f, 0.f, 0.f, 0.f};
  v8f acc[2][4];
  #pragma unroll
  for (int i = 0; i < 2; ++i)
    #pragma unroll
    for (int j = 0; j < 4; ++j) acc[i][j] = zerov;

  for (int k0 = 0; k0 < p.K; k0 += 32) {
    // ---------------- stage A tile (64 x 32) -----------------------------
    if (A16K) {
      // f16 source, K-contiguous: one 16B load/store per thread (8 halves)
      const int r  = tid >> 2;                 // 0..63
      const int kq = (tid & 3) << 3;           // 0,8,16,24
      const _Float16* src = (const _Float16*)p.Aptr + aoff0 +
                            (bm + r) * p.lda_m + (long)(k0 + kq);
      if (k0 + 32 < p.K) __builtin_prefetch(src + 32, 0, 1);
      *(uint4*)&As[r * 32 + kq] = *(const uint4*)src;
    } else {
      // f32 source, M-contiguous (q-path), optional per-K bias
      #pragma unroll
      for (int j = 0; j < 2; ++j) {
        const int q  = tid + j * 256;          // 0..511 float4 groups
        const int k  = q >> 4;                 // 0..31
        const int m4 = (q & 15) << 2;          // 0..60
        const float* src = (const float*)p.Aptr + aoff0 + (bm + m4) +
                           (long)(k0 + k) * p.lda_k;
        const float4 v = *(const float4*)src;
        const float ab = abp ? abp[k0 + k] : 0.f;
        As[(m4 + 0) * 32 + k] = (_Float16)(v.x + ab);
        As[(m4 + 1) * 32 + k] = (_Float16)(v.y + ab);
        As[(m4 + 2) * 32 + k] = (_Float16)(v.z + ab);
        As[(m4 + 3) * 32 + k] = (_Float16)(v.w + ab);
      }
    }
    // ---------------- stage B tile (32 x 256) ----------------------------
    if (BNCONT) {
      #pragma unroll
      for (int j = 0; j < 8; ++j) {
        const int q  = tid + j * 256;          // 0..2047 groups of 4 (along n)
        const int k  = q >> 6;                 // 0..31
        const int n4 = (q & 63) << 2;          // 0..252
        float f0 = 0.f, f1 = 0.f, f2 = 0.f, f3 = 0.f;
        if (COMP) {
          const long tt = nt0 + n4 + p.shift;
          const float* src = (const float*)p.Bptr + boff0 +
                             (long)(k0 + k) * p.ldb_k + bi * p.ldb_bat + tt;
          if (tt >= 0 && tt + 3 < p.Tperiod) {          // interior fast path
            f0 = src[0]; f1 = src[1]; f2 = src[2]; f3 = src[3];
          } else {                                       // shifted boundary
            if (tt + 0 >= 0 && tt + 0 < p.Tperiod) f0 = src[0];
            if (tt + 1 >= 0 && tt + 1 < p.Tperiod) f1 = src[1];
            if (tt + 2 >= 0 && tt + 2 < p.Tperiod) f2 = src[2];
            if (tt + 3 >= 0 && tt + 3 < p.Tperiod) f3 = src[3];
          }
        } else {
          const long n = bn + n4;
          const float* src = (const float*)p.Bptr + boff0 +
                             (long)(k0 + k) * p.ldb_k + n;
          if (n + 3 < p.N) { f0 = src[0]; f1 = src[1]; f2 = src[2]; f3 = src[3]; }
          else {
            if (n + 0 < p.N) f0 = src[0];
            if (n + 1 < p.N) f1 = src[1];
            if (n + 2 < p.N) f2 = src[2];
            if (n + 3 < p.N) f3 = src[3];
          }
        }
        if (BRELU) {
          f0 = fmaxf(f0, 0.f); f1 = fmaxf(f1, 0.f);
          f2 = fmaxf(f2, 0.f); f3 = fmaxf(f3, 0.f);
        }
        Bs[(n4 + 0) * 32 + k] = (_Float16)f0;
        Bs[(n4 + 1) * 32 + k] = (_Float16)f1;
        Bs[(n4 + 2) * 32 + k] = (_Float16)f2;
        Bs[(n4 + 3) * 32 + k] = (_Float16)f3;
      }
    } else {
      // f32 source, K-contiguous (v-path): float4 load -> packed b64 store
      #pragma unroll
      for (int j = 0; j < 8; ++j) {
        const int q  = tid + j * 256;          // 0..2047
        const int n  = q >> 3;                 // 0..255
        const int k4 = (q & 7) << 2;           // 0,4,...,28
        float4 v = make_float4(0.f, 0.f, 0.f, 0.f);
        if (bn + n < p.N)
          v = *(const float4*)((const float*)p.Bptr + boff0 +
                               (bn + n) * p.ldb_n + (long)(k0 + k4));
        union { _Float16 h[4]; unsigned long long u; } pk;
        pk.h[0] = (_Float16)v.x; pk.h[1] = (_Float16)v.y;
        pk.h[2] = (_Float16)v.z; pk.h[3] = (_Float16)v.w;
        *(unsigned long long*)&Bs[n * 32 + k4] = pk.u;
      }
    }
    __syncthreads();

    // ---------------- WMMA: 2x4 tiles of 16x16x32 ------------------------
    #pragma unroll
    for (int im = 0; im < 2; ++im) {
      union { v16h v; unsigned u[8]; } fa;
      const int mrow = wm * 32 + im * 16 + lr;
      #pragma unroll
      for (int d = 0; d < 8; ++d) {
        // A 16x32 f16, contiguous k order used here (matches B): lane-half 0 holds K=0..15, half 1 K=16..31
        const int kk = 2 * d + half * 16;
        fa.u[d] = *(const unsigned*)&As[mrow * 32 + kk];
      }
      #pragma unroll
      for (int in = 0; in < 4; ++in) {
        union { v16h v; unsigned u[8]; } fb;
        const int nrow = wn * 64 + in * 16 + lr;
        #pragma unroll
        for (int d = 0; d < 8; ++d) {
          // B 32x16 f16, contiguous k order used here (matches A): lane-half 0 holds K=0..15, half 1 K=16..31
          const int kk = 2 * d + half * 16;
          fb.u[d] = *(const unsigned*)&Bs[nrow * 32 + kk];
        }
        acc[im][in] = __builtin_amdgcn_wmma_f32_16x16x32_f16(
            false, fa.v, false, fb.v, (short)0, acc[im][in], false, false);
      }
    }
    __syncthreads();
  }

  // ---------------- epilogue (M always a multiple of 64) -----------------
  float* Cb = p.Cptr + coff0;
  #pragma unroll
  for (int im = 0; im < 2; ++im)
  #pragma unroll
  for (int in = 0; in < 4; ++in) {
    #pragma unroll
    for (int j = 0; j < 8; ++j) {
      const long m  = bm + wm * 32 + im * 16 + j + 8 * half;
      const long no = wn * 64 + in * 16 + lr;        // local n
      const long n  = bn + no;
      if (n < p.N) {
        long coff;
        if (COMP) coff = m * p.ldc_m + bi * p.ldc_bat + (nt0 + no) * p.ldc_n;
        else      coff = m * p.ldc_m + n * p.ldc_n;
        float v = p.alpha * acc[im][in][j];
        if (p.bias_m) v += p.bias_m[m];
        if (p.beta != 0.f) v += p.beta * Cb[coff];
        if (p.add_src) {
          const long aoff = COMP
              ? m * p.add_ldm + bi * p.add_bat + (nt0 + no) * p.add_ldn
              : m * p.add_ldm + n * p.add_ldn;
          v += p.add_scale * p.add_src[aoff];
        }
        Cb[coff] = v;
      }
    }
  }
}

// ---------------------------------------------------------------------------
// Support kernels
// ---------------------------------------------------------------------------
__global__ __launch_bounds__(256) void layernorm_kernel(
    const float* __restrict__ x, const float* __restrict__ g,
    const float* __restrict__ b, float* __restrict__ y) {
  const int n = blockIdx.x * blockDim.x + threadIdx.x;  // over (b, t)
  if (n >= kN) return;
  const int bb = n / kT, t = n % kT;
  const float* xp = x + (long)bb * kC * kT + t;
  float s = 0.f, s2 = 0.f;
  for (int c = 0; c < kC; ++c) { const float v = xp[(long)c * kT]; s += v; s2 += v * v; }
  const float m = s * (1.f / kC);
  const float var = s2 * (1.f / kC) - m * m;
  const float inv = rsqrtf(var + kEPS);
  float* yp = y + (long)bb * kC * kT + t;
  for (int c = 0; c < kC; ++c)
    yp[(long)c * kT] = (xp[(long)c * kT] - m) * inv * g[c] + b[c];
}

__global__ __launch_bounds__(256) void posembed_kernel(float* __restrict__ pe) {
  const int idx = blockIdx.x * blockDim.x + threadIdx.x;
  if (idx >= kC * kS2) return;
  const int c = idx / kS2, j = idx % kS2;
  const int i2 = (c >> 1) * 2;
  const float f = 256.f * powf(10000.f, -(float)i2);
  float ang;
  if (j < kT) ang = (float)(kT - 1 - j) * f;       // reversed pos_pos
  else        ang = -(float)(j - kT + 1) * f;      // pos_neg[:,1:]
  pe[idx] = (c & 1) ? cosf(ang) : sinf(ang);
}

__global__ __launch_bounds__(256) void softmax_kernel(
    const float* __restrict__ ac, const float* __restrict__ bd,
    const int* __restrict__ mask, _Float16* __restrict__ probs) {
  const int t = blockIdx.x;
  const int z = blockIdx.y;           // (b,h)
  const int b = z / kH;
  const int tid = threadIdx.x;
  _Float16* prow = probs + ((long)z * kT + t) * kT;
  __shared__ float red[256];
  if (mask[b * kT + t] == 0) {        // fully-masked query row -> zeros
    for (int s = tid; s < kT; s += 256) prow[s] = (_Float16)0.f;
    return;
  }
  const float scale = 0.08838834764831845f;  // 1/sqrt(128)
  const float* acr = ac + ((long)z * kT + t) * kT;
  // rel_shift folded in: out[t,s] = bd[t, s - t + T - 1]
  const float* bdr = bd + ((long)z * kT + t) * (long)kS2 + (kT - 1 - t);
  float sc[3]; float mx = -3.4e38f;
  #pragma unroll
  for (int i = 0; i < 3; ++i) {
    const int s = tid + i * 256;
    const float v = (acr[s] + bdr[s]) * scale;
    sc[i] = v; mx = fmaxf(mx, v);
  }
  red[tid] = mx; __syncthreads();
  for (int o = 128; o > 0; o >>= 1) { if (tid < o) red[tid] = fmaxf(red[tid], red[tid + o]); __syncthreads(); }
  mx = red[0]; __syncthreads();
  float sum = 0.f;
  #pragma unroll
  for (int i = 0; i < 3; ++i) { sc[i] = expf(sc[i] - mx); sum += sc[i]; }
  red[tid] = sum; __syncthreads();
  for (int o = 128; o > 0; o >>= 1) { if (tid < o) red[tid] += red[tid + o]; __syncthreads(); }
  const float inv = 1.f / red[0];
  #pragma unroll
  for (int i = 0; i < 3; ++i) prow[tid + i * 256] = (_Float16)(sc[i] * inv);
}

__global__ __launch_bounds__(256) void glu_kernel(
    const float* __restrict__ in, float* __restrict__ out) {
  const long idx = (long)blockIdx.x * blockDim.x + threadIdx.x;
  if (idx >= (long)kB * kC * kT) return;
  const long b = idx / ((long)kC * kT);
  const long r = idx % ((long)kC * kT);
  const long c = r / kT, t = r % kT;
  const long base = b * (long)(2 * kC) * kT;
  const float a = in[base + c * kT + t];
  const float g = in[base + (c + kC) * kT + t];
  out[idx] = a / (1.f + expf(-g));
}

__global__ __launch_bounds__(256) void bnstats_kernel(
    const float* __restrict__ x, float* __restrict__ mu, float* __restrict__ var) {
  const int c = blockIdx.x, tid = threadIdx.x;
  __shared__ float s1[256], s2[256];
  float a = 0.f, b2 = 0.f;
  for (int n = tid; n < kB * kT; n += 256) {
    const int bb = n / kT, t = n % kT;
    const float v = x[(long)bb * kC * kT + (long)c * kT + t];
    a += v; b2 += v * v;
  }
  s1[tid] = a; s2[tid] = b2; __syncthreads();
  for (int o = 128; o > 0; o >>= 1) {
    if (tid < o) { s1[tid] += s1[tid + o]; s2[tid] += s2[tid + o]; }
    __syncthreads();
  }
  if (tid == 0) {
    const float m = s1[0] / (float)(kB * kT);
    mu[c] = m; var[c] = s2[0] / (float)(kB * kT) - m * m;
  }
}

__global__ __launch_bounds__(256) void bnsilu_kernel(
    float* __restrict__ x, const float* __restrict__ mu, const float* __restrict__ var,
    const float* __restrict__ g, const float* __restrict__ b) {
  const long idx = (long)blockIdx.x * blockDim.x + threadIdx.x;
  if (idx >= (long)kB * kC * kT) return;
  const int c = (int)((idx / kT) % kC);
  float v = (x[idx] - mu[c]) * rsqrtf(var[c] + kEPS) * g[c] + b[c];
  x[idx] = v / (1.f + expf(-v));   // x * sigmoid(x)
}

__global__ __launch_bounds__(256) void maskmul_kernel(
    float* __restrict__ x, const int* __restrict__ mask) {
  const long idx = (long)blockIdx.x * blockDim.x + threadIdx.x;
  if (idx >= (long)kB * kC * kT) return;
  const long b = idx / ((long)kC * kT);
  const long t = idx % kT;
  x[idx] *= (float)mask[b * kT + t];
}

// Repack conv/linear weights (O, Ci, KW) f32 -> f16 as [KW][O][Ci]
__global__ __launch_bounds__(256) void wconv_kernel(
    const float* __restrict__ src, _Float16* __restrict__ dst,
    int total, int Ci, int KW, long OCi) {
  const int idx = blockIdx.x * blockDim.x + threadIdx.x;
  if (idx >= total) return;
  const int j = idx % KW;
  const int c = (idx / KW) % Ci;
  const int o = idx / (KW * Ci);
  dst[(long)j * OCi + (long)o * Ci + c] = (_Float16)src[idx];
}

// ---------------------------------------------------------------------------
// Host orchestration
// ---------------------------------------------------------------------------
template <bool A16K, bool BNCONT, bool COMP, bool BRELU>
static void launch_gemm_t(const GP& g, int batches, hipStream_t stream) {
  dim3 grid((g.N + 255) / 256, g.M / 64, batches);
  gemm_wmma_kernel<A16K, BNCONT, COMP, BRELU><<<grid, dim3(256), 0, stream>>>(g);
}

extern "C" void kernel_launch(void* const* d_in, const int* in_sizes, int n_in,
                              void* d_out, int out_size, void* d_ws, size_t ws_size,
                              hipStream_t stream) {
  (void)in_sizes; (void)out_size; (void)ws_size;

  // ---- workspace carve-out ---------------------------------------------
  char* wp = (char*)d_ws;
  auto alloc = [&](size_t bytes) -> void* {
    void* p = (void*)wp; wp += (bytes + 255) & ~(size_t)255; return p;
  };
  const size_t xbytes = sizeof(float) * (size_t)kB * kC * kT;
  float* xA     = (float*)alloc(xbytes);
  float* xB     = (float*)alloc(xbytes);
  float* xln    = (float*)alloc(xbytes);
  float* midb   = (float*)alloc(sizeof(float) * (size_t)kB * kMID * kT);
  float* qb     = (float*)alloc(xbytes);
  float* kb     = (float*)alloc(xbytes);
  float* vb     = (float*)alloc(xbytes);
  float* attnb  = (float*)alloc(xbytes);
  float* buf512 = (float*)alloc(sizeof(float) * (size_t)kB * 2 * kC * kT);
  float* glb    = (float*)alloc(xbytes);
  float* dbuf   = (float*)alloc(xbytes);
  float* peraw  = (float*)alloc(sizeof(float) * (size_t)kC * kS2);
  float* peproj = (float*)alloc(sizeof(float) * (size_t)kC * kS2);
  float* acws   = (float*)alloc(sizeof(float) * (size_t)kB * kH * kT * kT);
  float* bdws   = (float*)alloc(sizeof(float) * (size_t)kB * kH * kT * kS2);
  _Float16* probs  = (_Float16*)alloc(sizeof(_Float16) * (size_t)kB * kH * kT * kT);
  float* bnmu   = (float*)alloc(sizeof(float) * kC);
  float* bnvar  = (float*)alloc(sizeof(float) * kC);
  _Float16* harena = (_Float16*)alloc(sizeof(_Float16) * (size_t)6 * kMID * kC);

  const int* maskp = (const int*)d_in[1];
  const float* P[216];
  const int nP = (n_in - 2 < 216) ? (n_in - 2) : 216;
  for (int i = 0; i < nP; ++i) P[i] = (const float*)d_in[2 + i];

  float* xcur = xA;
  float* xnext = xB;
  auto ew_grid = [&](long total) { return dim3((unsigned)((total + 255) / 256)); };

  // x0 <- input; precompute positional embedding (layer-invariant)
  hipMemcpyAsync(xA, d_in[0], xbytes, hipMemcpyDeviceToDevice, stream);
  posembed_kernel<<<ew_grid((long)kC * kS2), 256, 0, stream>>>(peraw);

  // ------------------------------- FFN ----------------------------------
  auto do_ffn = [&](int pb) {
    const float *ln_g = P[pb], *ln_b = P[pb + 1], *w1 = P[pb + 2],
                *b1 = P[pb + 3], *w2 = P[pb + 4], *b2 = P[pb + 5];
    _Float16* hw1 = harena;
    _Float16* hw2 = harena + (long)3 * kMID * kC;
    int t1 = kMID * kC * 3;
    wconv_kernel<<<ew_grid(t1), 256, 0, stream>>>(w1, hw1, t1, kC, 3, (long)kMID * kC);
    int t2 = kC * kMID * 3;
    wconv_kernel<<<ew_grid(t2), 256, 0, stream>>>(w2, hw2, t2, kMID, 3, (long)kC * kMID);
    layernorm_kernel<<<ew_grid(kN), 256, 0, stream>>>(xcur, ln_g, ln_b, xln);
    // conv1 (C->MID, k=3) as 3 shifted accumulated GEMMs
    for (int j = 0; j < 3; ++j) {
      GP g{};
      g.Aptr = hw1 + (long)j * kMID * kC; g.lda_m = kC; g.lda_k = 1;
      g.Bptr = xln; g.ldb_k = kT; g.ldb_bat = (long)kC * kT; g.ldb_n = 1;
      g.Cptr = midb; g.ldc_m = kT; g.ldc_bat = (long)kMID * kT; g.ldc_n = 1;
      g.M = kMID; g.N = kN; g.K = kC; g.nbh = 1; g.Tperiod = kT; g.shift = j - 1;
      g.alpha = 1.f; g.beta = (j == 0) ? 0.f : 1.f;
      g.bias_m = (j == 0) ? b1 : nullptr;
      launch_gemm_t<true, true, true, false>(g, 1, stream);
    }
    // conv2 (MID->C, k=3); ReLU applied on B-load; final: x + 0.5*(...)
    for (int j = 0; j < 3; ++j) {
      GP g{};
      g.Aptr = hw2 + (long)j * kC * kMID; g.lda_m = kMID; g.lda_k = 1;
      g.Bptr = midb; g.ldb_k = kT; g.ldb_bat = (long)kMID * kT; g.ldb_n = 1;
      g.Cptr = xnext; g.ldc_m = kT; g.ldc_bat = (long)kC * kT; g.ldc_n = 1;
      g.M = kC; g.N = kN; g.K = kMID; g.nbh = 1; g.Tperiod = kT; g.shift = j - 1;
      if (j == 0) { g.alpha = 1.f; g.beta = 0.f; g.bias_m = b2; }
      else if (j == 1) { g.alpha = 1.f; g.beta = 1.f; }
      else {
        g.alpha = 0.5f; g.beta = 0.5f;
        g.add_src = xcur; g.add_scale = 1.f;
        g.add_ldm = kT; g.add_bat = (long)kC * kT; g.add_ldn = 1;
      }
      launch_gemm_t<true, true, true, true>(g, 1, stream);
    }
    float* tmp = xcur; xcur = xnext; xnext = tmp;
  };

  // ---------------------------- Attention --------------------------------
  auto do_attn = [&](int pb) {
    const float *ln_g = P[pb], *ln_b = P[pb + 1], *wq = P[pb + 2], *bq = P[pb + 3],
                *wk = P[pb + 4], *bk = P[pb + 5], *wv = P[pb + 6], *bv = P[pb + 7],
                *wpos = P[pb + 8], *bpos = P[pb + 9], *wout = P[pb + 10],
                *bout = P[pb + 11], *bias_u = P[pb + 12], *bias_v = P[pb + 13];
    _Float16 *hq = harena, *hk = hq + (long)kC * kC, *hv = hk + (long)kC * kC,
             *hpos = hv + (long)kC * kC, *hout = hpos + (long)kC * kC;
    const int tw = kC * kC;
    wconv_kernel<<<ew_grid(tw), 256, 0, stream>>>(wq,   hq,   tw, kC, 1, (long)kC * kC);
    wconv_kernel<<<ew_grid(tw), 256, 0, stream>>>(wk,   hk,   tw, kC, 1, (long)kC * kC);
    wconv_kernel<<<ew_grid(tw), 256, 0, stream>>>(wv,   hv,   tw, kC, 1, (long)kC * kC);
    wconv_kernel<<<ew_grid(tw), 256, 0, stream>>>(wpos, hpos, tw, kC, 1, (long)kC * kC);
    wconv_kernel<<<ew_grid(tw), 256, 0, stream>>>(wout, hout, tw, kC, 1, (long)kC * kC);
    layernorm_kernel<<<ew_grid(kN), 256, 0, stream>>>(xcur, ln_g, ln_b, xln);

    auto proj = [&](_Float16* W, const float* bias, float* out) {
      GP g{};
      g.Aptr = W; g.lda_m = kC; g.lda_k = 1;
      g.Bptr = xln; g.ldb_k = kT; g.ldb_bat = (long)kC * kT; g.ldb_n = 1;
      g.Cptr = out; g.ldc_m = kT; g.ldc_bat = (long)kC * kT; g.ldc_n = 1;
      g.M = kC; g.N = kN; g.K = kC; g.nbh = 1; g.Tperiod = kT;
      g.alpha = 1.f; g.bias_m = bias;
      launch_gemm_t<true, true, true, false>(g, 1, stream);
    };
    proj(hq, bq, qb); proj(hk, bk, kb); proj(hv, bv, vb);

    {   // positional projection: (C x C) @ (C x S2)
      GP g{};
      g.Aptr = hpos; g.lda_m = kC; g.lda_k = 1;
      g.Bptr = peraw; g.ldb_k = kS2; g.ldb_n = 1;
      g.Cptr = peproj; g.ldc_m = kS2; g.ldc_n = 1;
      g.M = kC; g.N = kS2; g.K = kC; g.nbh = 1;
      g.alpha = 1.f; g.bias_m = bpos;
      launch_gemm_t<true, true, false, false>(g, 1, stream);
    }
    {   // ac = (q + bias_u) @ k^T   per (b,h)
      GP g{};
      g.Aptr = qb; g.lda_m = 1; g.lda_k = kT;
      g.batA_b = (long)kC * kT; g.batA_h = (long)kHS * kT;
      g.abias = bias_u; g.abias_h = kHS;
      g.Bptr = kb; g.ldb_k = kT; g.ldb_n = 1;
      g.batB_b = (long)kC * kT; g.batB_h = (long)kHS * kT;
      g.Cptr = acws; g.ldc_m = kT; g.ldc_n = 1;
      g.batC_b = (long)kH * kT * kT; g.batC_h = (long)kT * kT;
      g.M = kT; g.N = kT; g.K = kHS; g.nbh = kH; g.alpha = 1.f;
      launch_gemm_t<false, true, false, false>(g, kB * kH, stream);
    }
    {   // bd = (q + bias_v) @ pe^T  per (b,h); rel_shift folded into softmax
      GP g{};
      g.Aptr = qb; g.lda_m = 1; g.lda_k = kT;
      g.batA_b = (long)kC * kT; g.batA_h = (long)kHS * kT;
      g.abias = bias_v; g.abias_h = kHS;
      g.Bptr = peproj; g.ldb_k = kS2; g.ldb_n = 1;
      g.batB_b = 0; g.batB_h = (long)kHS * kS2;
      g.Cptr = bdws; g.ldc_m = kS2; g.ldc_n = 1;
      g.batC_b = (long)kH * kT * kS2; g.batC_h = (long)kT * kS2;
      g.M = kT; g.N = kS2; g.K = kHS; g.nbh = kH; g.alpha = 1.f;
      launch_gemm_t<false, true, false, false>(g, kB * kH, stream);
    }
    softmax_kernel<<<dim3(kT, kB * kH), 256, 0, stream>>>(acws, bdws, maskp, probs);
    {   // out = probs @ v  per (b,h), written back into (b, c, t) layout
      GP g{};
      g.Aptr = probs; g.lda_m = kT; g.lda_k = 1;
      g.batA_b = (long)kH * kT * kT; g.batA_h = (long)kT * kT;
      g.Bptr = vb; g.ldb_k = 1; g.ldb_n = kT;
      g.batB_b = (long)kC * kT; g.batB_h = (long)kHS * kT;
      g.Cptr = attnb; g.ldc_m = 1; g.ldc_n = kT;
      g.batC_b = (long)kC * kT; g.batC_h = (long)kHS * kT;
      g.M = kT; g.N = kHS; g.K = kT; g.nbh = kH; g.alpha = 1.f;
      launch_gemm_t<true, false, false, false>(g, kB * kH, stream);
    }
    {   // final projection + bias + residual
      GP g{};
      g.Aptr = hout; g.lda_m = kC; g.lda_k = 1;
      g.Bptr = attnb; g.ldb_k = kT; g.ldb_bat = (long)kC * kT; g.ldb_n = 1;
      g.Cptr = xnext; g.ldc_m = kT; g.ldc_bat = (long)kC * kT; g.ldc_n = 1;
      g.M = kC; g.N = kN; g.K = kC; g.nbh = 1; g.Tperiod = kT;
      g.alpha = 1.f; g.bias_m = bout;
      g.add_src = xcur; g.add_scale = 1.f;
      g.add_ldm = kT; g.add_bat = (long)kC * kT; g.add_ldn = 1;
      launch_gemm_t<true, true, true, false>(g, 1, stream);
    }
    float* tmp = xcur; xcur = xnext; xnext = tmp;
  };

  // --------------------------- Conv module --------------------------------
  auto do_conv = [&](int pb) {
    const float *ln_g = P[pb], *ln_b = P[pb + 1], *pw1_w = P[pb + 2], *pw1_b = P[pb + 3],
                *dw_w = P[pb + 4], *dw_b = P[pb + 5], *bn_g = P[pb + 6], *bn_b = P[pb + 7],
                *pw2_w = P[pb + 8], *pw2_b = P[pb + 9];
    _Float16* hpw1 = harena;                          // 512*256
    _Float16* hdw  = hpw1 + (long)2 * kC * kC;        // 7*256*256
    _Float16* hpw2 = hdw + (long)7 * kC * kC;         // 256*256
    int t1 = 2 * kC * kC;
    wconv_kernel<<<ew_grid(t1), 256, 0, stream>>>(pw1_w, hpw1, t1, kC, 1, (long)2 * kC * kC);
    int t2 = kC * kC * 7;
    wconv_kernel<<<ew_grid(t2), 256, 0, stream>>>(dw_w, hdw, t2, kC, 7, (long)kC * kC);
    int t3 = kC * kC;
    wconv_kernel<<<ew_grid(t3), 256, 0, stream>>>(pw2_w, hpw2, t3, kC, 1, (long)kC * kC);
    layernorm_kernel<<<ew_grid(kN), 256, 0, stream>>>(xcur, ln_g, ln_b, xln);
    {   // pointwise 1: C -> 2C
      GP g{};
      g.Aptr = hpw1; g.lda_m = kC; g.lda_k = 1;
      g.Bptr = xln; g.ldb_k = kT; g.ldb_bat = (long)kC * kT; g.ldb_n = 1;
      g.Cptr = buf512; g.ldc_m = kT; g.ldc_bat = (long)2 * kC * kT; g.ldc_n = 1;
      g.M = 2 * kC; g.N = kN; g.K = kC; g.nbh = 1; g.Tperiod = kT;
      g.alpha = 1.f; g.bias_m = pw1_b;
      launch_gemm_t<true, true, true, false>(g, 1, stream);
    }
    glu_kernel<<<ew_grid((long)kB * kC * kT), 256, 0, stream>>>(buf512, glb);
    // dense conv k=7 as 7 shifted accumulated GEMMs
    for (int j = 0; j < 7; ++j) {
      GP g{};
      g.Aptr = hdw + (long)j * kC * kC; g.lda_m = kC; g.lda_k = 1;
      g.Bptr = glb; g.ldb_k = kT; g.ldb_bat = (long)kC * kT; g.ldb_n = 1;
      g.Cptr = dbuf; g.ldc_m = kT; g.ldc_bat = (long)kC * kT; g.ldc_n = 1;
      g.M = kC; g.N = kN; g.K = kC; g.nbh = 1; g.Tperiod = kT; g.shift = j - 3;
      g.alpha = 1.f; g.beta = (j == 0) ? 0.f : 1.f;
      g.bias_m = (j == 0) ? dw_b : nullptr;
      launch_gemm_t<true, true, true, false>(g, 1, stream);
    }
    bnstats_kernel<<<dim3(kC), 256, 0, stream>>>(dbuf, bnmu, bnvar);
    bnsilu_kernel<<<ew_grid((long)kB * kC * kT), 256, 0, stream>>>(dbuf, bnmu, bnvar, bn_g, bn_b);
    {   // pointwise 2 + residual
      GP g{};
      g.Aptr = hpw2; g.lda_m = kC; g.lda_k = 1;
      g.Bptr = dbuf; g.ldb_k = kT; g.ldb_bat = (long)kC * kT; g.ldb_n = 1;
      g.Cptr = xnext; g.ldc_m = kT; g.ldc_bat = (long)kC * kT; g.ldc_n = 1;
      g.M = kC; g.N = kN; g.K = kC; g.nbh = 1; g.Tperiod = kT;
      g.alpha = 1.f; g.bias_m = pw2_b;
      g.add_src = xcur; g.add_scale = 1.f;
      g.add_ldm = kT; g.add_bat = (long)kC * kT; g.add_ldn = 1;
      launch_gemm_t<true, true, true, false>(g, 1, stream);
    }
    float* tmp = xcur; xcur = xnext; xnext = tmp;
  };

  // ----------------------------- layers -----------------------------------
  for (int l = 0; l < kL; ++l) {
    const int base = l * 36;
    do_ffn(base + 0);        // ffn1
    do_attn(base + 6);       // attention
    do_conv(base + 20);      // conv module
    do_ffn(base + 30);       // ffn2
    maskmul_kernel<<<ew_grid((long)kB * kC * kT), 256, 0, stream>>>(xcur, maskp);
  }
  hipMemcpyAsync(d_out, xcur, xbytes, hipMemcpyDeviceToDevice, stream);
}
